// SelfAttentionBasicv2_72249939853362
// MI455X (gfx1250) — matrix operand order repaired and checked by reading, not compiled, not run on hardware
//
#include <hip/hip_runtime.h>

#define SEQ 4096
#define DMODEL 1024

typedef __attribute__((ext_vector_type(16))) __bf16 v16bf;
typedef __attribute__((ext_vector_type(8)))  float  v8f;
typedef __attribute__((ext_vector_type(16))) unsigned short v16u;
typedef __attribute__((ext_vector_type(8)))  unsigned short v8u;

// hardware RNE convert (v_cvt_*bf16_f32) instead of software rounding
__device__ __forceinline__ unsigned short f2bf(float f) {
  return __builtin_bit_cast(unsigned short, (__bf16)f);
}

__device__ __forceinline__ v16bf mkfrag(v8u lo, v8u hi) {
  v16u v = __builtin_shufflevector(lo, hi, 0,1,2,3,4,5,6,7,8,9,10,11,12,13,14,15);
  return __builtin_bit_cast(v16bf, v);
}

__device__ __forceinline__ v8f wmma_bf16(v16bf a, v16bf b, v8f c) {
  return __builtin_amdgcn_wmma_f32_16x16x32_bf16(false, a, false, b, (short)0, c,
                                                 false, false);
}

// ---------------------------------------------------------------------------
// Kernel A: C[M,N] = X[M,K] @ W[N,K]^T  -> bf16 output (optionally scaled)
// Block tile 128(M) x 64(N), 128 threads = 4 waves (2x2); each wave owns a
// 64x32 region = 8 WMMA tiles per 32-wide K chunk (4 A-frags x 2 B-frags).
// ---------------------------------------------------------------------------
__global__ __launch_bounds__(128)
void qkv_gemm(const float* __restrict__ X, const float* __restrict__ W,
              unsigned short* __restrict__ Out, float outScale) {
  __shared__ unsigned short Xs[128][40];   // 128 rows x 32 k (pad 40)
  __shared__ unsigned short Ws[64][40];

  const int tid  = threadIdx.x;
  const int lane = tid & 31, wave = tid >> 5;
  const int h = lane >> 4, ln = lane & 15;
  const int wm = wave >> 1, wn = wave & 1;
  const int tileM = blockIdx.y * 128, tileN = blockIdx.x * 64;

  v8f acc[4][2] = {};

  for (int kc = 0; kc < DMODEL / 32; ++kc) {
    // prefetch next K chunk (one 128B line per row; fire-and-forget)
    {
      int kn = (kc + 1) & (DMODEL / 32 - 1);
      __builtin_prefetch(X + (size_t)(tileM + tid) * DMODEL + kn * 32, 0, 3);
      if (tid < 64)
        __builtin_prefetch(W + (size_t)(tileN + tid) * DMODEL + kn * 32, 0, 3);
    }
#pragma unroll
    for (int i = 0; i < 8; ++i) {
      int idx = i * 128 + tid;            // 0..1023 -> 128 rows x 8 float4
      int r = idx >> 3, c4 = (idx & 7) * 4;
      float4 xf = *(const float4*)(X + (size_t)(tileM + r) * DMODEL + kc * 32 + c4);
      Xs[r][c4 + 0] = f2bf(xf.x); Xs[r][c4 + 1] = f2bf(xf.y);
      Xs[r][c4 + 2] = f2bf(xf.z); Xs[r][c4 + 3] = f2bf(xf.w);
    }
#pragma unroll
    for (int i = 0; i < 4; ++i) {
      int idx = i * 128 + tid;            // 0..511 -> 64 rows x 8 float4
      int r = idx >> 3, c4 = (idx & 7) * 4;
      float4 wf = *(const float4*)(W + (size_t)(tileN + r) * DMODEL + kc * 32 + c4);
      Ws[r][c4 + 0] = f2bf(wf.x); Ws[r][c4 + 1] = f2bf(wf.y);
      Ws[r][c4 + 2] = f2bf(wf.z); Ws[r][c4 + 3] = f2bf(wf.w);
    }
    __syncthreads();

    v16bf a[4], b[2];
#pragma unroll
    for (int t = 0; t < 4; ++t) {
      const unsigned short* xr = &Xs[wm * 64 + t * 16 + ln][0];
      a[t] = mkfrag(*(const v8u*)(xr + h * 8), *(const v8u*)(xr + 16 + h * 8));
    }
#pragma unroll
    for (int t = 0; t < 2; ++t) {
      const unsigned short* wr = &Ws[wn * 32 + t * 16 + ln][0];
      b[t] = mkfrag(*(const v8u*)(wr + h * 8), *(const v8u*)(wr + 16 + h * 8));
    }
#pragma unroll
    for (int ti = 0; ti < 4; ++ti)
#pragma unroll
      for (int tj = 0; tj < 2; ++tj)
        acc[ti][tj] = wmma_bf16(a[ti], b[tj], acc[ti][tj]);
    __syncthreads();
  }

#pragma unroll
  for (int ti = 0; ti < 4; ++ti)
#pragma unroll
    for (int tj = 0; tj < 2; ++tj)
#pragma unroll
      for (int r = 0; r < 8; ++r) {
        int row = tileM + wm * 64 + ti * 16 + h * 8 + r;
        int col = tileN + wn * 32 + tj * 16 + ln;
        Out[(size_t)row * DMODEL + col] = f2bf(acc[ti][tj][r] * outScale);
      }
}

// ---------------------------------------------------------------------------
// Kernel B: flash attention. Block = 16 query rows, 128 threads (4 waves).
// Wave w: computes S for keys [kb*64+w*16, +16), then owns output features
// [w*256, (w+1)*256) for O += P@V. Online softmax stats shared via LDS.
// ---------------------------------------------------------------------------
__global__ __launch_bounds__(128)
void flash_attn(const unsigned short* __restrict__ Qb,
                const unsigned short* __restrict__ Kb,
                const unsigned short* __restrict__ Vb,
                float* __restrict__ Out) {
  __shared__ unsigned short VT[4][64][72];  // per-wave V^T slice: [feat][key]
  __shared__ unsigned short Pl[16][72];     // P tile [row][key 0..63]
  __shared__ float redmax[4][16];
  __shared__ float redsum[4][16];

  const int tid  = threadIdx.x;
  const int lane = tid & 31, wave = tid >> 5;
  const int h = lane >> 4, ln = lane & 15;
  const int qBase = blockIdx.x * 16;

  v8f accO[16] = {};                 // 4 sub-slices x 4 col tiles (16x16 each)
  float m[8], l[8];
#pragma unroll
  for (int r = 0; r < 8; ++r) { m[r] = -1e30f; l[r] = 0.f; }

  const unsigned short* qrow = Qb + (size_t)(qBase + ln) * DMODEL;

  for (int kb = 0; kb < SEQ / 64; ++kb) {
    // prefetch next key block of K and V (64 rows x 2KB = 16 lines/row)
    {
      int nb = (kb + 1) & (SEQ / 64 - 1);
      const size_t base = (size_t)nb * 64 * DMODEL;
#pragma unroll
      for (int i = 0; i < 8; ++i) {
        int id = i * 128 + tid;           // 0..1023
        int row = id >> 4, seg = id & 15; // 64 rows x 16 x 128B
        __builtin_prefetch(Kb + base + (size_t)row * DMODEL + seg * 64, 0, 3);
        __builtin_prefetch(Vb + base + (size_t)row * DMODEL + seg * 64, 0, 3);
      }
    }

    // ---- S tile (16 rows x 16 keys), accumulate over d=1024 ----
    v8f accS = {0.f, 0.f, 0.f, 0.f, 0.f, 0.f, 0.f, 0.f};
    const unsigned short* krow =
        Kb + (size_t)(kb * 64 + wave * 16 + ln) * DMODEL;
    for (int fc = 0; fc < DMODEL / 32; ++fc) {
      v16bf aQ = mkfrag(*(const v8u*)(qrow + fc * 32 + h * 8),
                        *(const v8u*)(qrow + fc * 32 + 16 + h * 8));
      v16bf bK = mkfrag(*(const v8u*)(krow + fc * 32 + h * 8),
                        *(const v8u*)(krow + fc * 32 + 16 + h * 8));
      accS = wmma_bf16(aQ, bK, accS);
    }

    // ---- online softmax over the 64-key block ----
    float rmax[8];
#pragma unroll
    for (int r = 0; r < 8; ++r) {
      float v = accS[r];
#pragma unroll
      for (int mk = 8; mk >= 1; mk >>= 1)
        v = fmaxf(v, __shfl_xor(v, mk, 32));
      rmax[r] = v;
    }
    if (ln == 0)
#pragma unroll
      for (int r = 0; r < 8; ++r) redmax[wave][h * 8 + r] = rmax[r];
    __syncthreads();

    float corr[8], mn[8];
#pragma unroll
    for (int r = 0; r < 8; ++r) {
      int row = h * 8 + r;
      float bm = fmaxf(fmaxf(redmax[0][row], redmax[1][row]),
                       fmaxf(redmax[2][row], redmax[3][row]));
      mn[r]   = fmaxf(m[r], bm);
      corr[r] = __expf(m[r] - mn[r]);
      m[r]    = mn[r];
    }

    float rsum[8];
#pragma unroll
    for (int r = 0; r < 8; ++r) {
      float p = __expf(accS[r] - mn[r]);
      Pl[h * 8 + r][wave * 16 + ln] = f2bf(p);
      float s = p;
#pragma unroll
      for (int mk = 8; mk >= 1; mk >>= 1)
        s += __shfl_xor(s, mk, 32);
      rsum[r] = s;
    }
    if (ln == 0)
#pragma unroll
      for (int r = 0; r < 8; ++r) redsum[wave][h * 8 + r] = rsum[r];

#pragma unroll
    for (int t = 0; t < 16; ++t)
#pragma unroll
      for (int r = 0; r < 8; ++r) accO[t][r] *= corr[r];
#pragma unroll
    for (int r = 0; r < 8; ++r) l[r] *= corr[r];
    __syncthreads();

#pragma unroll
    for (int r = 0; r < 8; ++r) {
      int row = h * 8 + r;
      l[r] += redsum[0][row] + redsum[1][row] + redsum[2][row] + redsum[3][row];
    }

    // P as A-fragments (16x64 -> two 16x32 chunks)
    const unsigned short* pr = &Pl[ln][0];
    v16bf pa0 = mkfrag(*(const v8u*)(pr + h * 8),      *(const v8u*)(pr + 16 + h * 8));
    v16bf pa1 = mkfrag(*(const v8u*)(pr + 32 + h * 8), *(const v8u*)(pr + 48 + h * 8));

    // ---- O += P @ V over this wave's 256-feature slice ----
#pragma unroll
    for (int s = 0; s < 4; ++s) {
      const int f0 = wave * 256 + s * 64;
#pragma unroll
      for (int it = 0; it < 16; ++it) {
        int id = it * 32 + lane;          // 0..511
        int k = id >> 3, g = id & 7;
        v8u vv = *(const v8u*)(Vb + (size_t)(kb * 64 + k) * DMODEL + f0 + g * 8);
#pragma unroll
        for (int e = 0; e < 8; ++e) VT[wave][g * 8 + e][k] = vv[e];
      }
      __syncthreads();
#pragma unroll
      for (int c = 0; c < 4; ++c) {
        const unsigned short* vr = &VT[wave][c * 16 + ln][0];
        v16bf b0 = mkfrag(*(const v8u*)(vr + h * 8),      *(const v8u*)(vr + 16 + h * 8));
        v16bf b1 = mkfrag(*(const v8u*)(vr + 32 + h * 8), *(const v8u*)(vr + 48 + h * 8));
        accO[s * 4 + c] = wmma_bf16(pa0, b0, accO[s * 4 + c]);
        accO[s * 4 + c] = wmma_bf16(pa1, b1, accO[s * 4 + c]);
      }
      __syncthreads();
    }
  }

  float inv[8];
#pragma unroll
  for (int r = 0; r < 8; ++r) inv[r] = 1.f / l[r];
#pragma unroll
  for (int s = 0; s < 4; ++s)
#pragma unroll
    for (int c = 0; c < 4; ++c)
#pragma unroll
      for (int r = 0; r < 8; ++r) {
        int row = qBase + h * 8 + r;
        int col = wave * 256 + s * 64 + c * 16 + ln;
        Out[(size_t)row * DMODEL + col] = accO[s * 4 + c][r] * inv[r];
      }
}

// ---------------------------------------------------------------------------
extern "C" void kernel_launch(void* const* d_in, const int* in_sizes, int n_in,
                              void* d_out, int out_size, void* d_ws, size_t ws_size,
                              hipStream_t stream) {
  const float* X  = (const float*)d_in[0];
  const float* Wq = (const float*)d_in[1];
  const float* Wk = (const float*)d_in[2];
  const float* Wv = (const float*)d_in[3];

  unsigned short* Qb = (unsigned short*)d_ws;                  // 8 MB each
  unsigned short* Kb = Qb + (size_t)SEQ * DMODEL;
  unsigned short* Vb = Kb + (size_t)SEQ * DMODEL;
  float* Out = (float*)d_out;

  dim3 gA(DMODEL / 64, SEQ / 128, 1);
  // fold score scale 1/sqrt(1024) = 1/32 (exact in bf16) into Q
  qkv_gemm<<<gA, 128, 0, stream>>>(X, Wq, Qb, 0.03125f);
  qkv_gemm<<<gA, 128, 0, stream>>>(X, Wk, Kb, 1.0f);
  qkv_gemm<<<gA, 128, 0, stream>>>(X, Wv, Vb, 1.0f);

  flash_attn<<<dim3(SEQ / 16), 128, 0, stream>>>(Qb, Kb, Vb, Out);
}
